// TCN_3410204033801
// MI455X (gfx1250) — matrix operand order repaired and checked by reading, not compiled, not run on hardware
//
#include <hip/hip_runtime.h>
#include <hip/hip_bf16.h>
#include <math.h>

// ---------------------------------------------------------------------------
// TCN + selection head for MI455X (gfx1250), wave32.
// Convs: WMMA bf16->f32, A = activations (M=time), B = transposed weights.
// Activations kept in [B][T][C] bf16 (native x layout) so LDS staging is
// contiguous 128B rows -> async global->LDS loads; all LDS fragment reads
// are 16B/32B aligned b128s (no sub-word repacking).
// ---------------------------------------------------------------------------

#define BN    2048
#define TN    252
#define CN    64
#define KK    3
#define LN    4
#define NOUTN 500
#define NSELN 20
#define UBF   0.1f
#define LBF   0.0f

typedef __bf16 v16bf __attribute__((ext_vector_type(16)));
typedef float  v8f   __attribute__((ext_vector_type(8)));

static __device__ __forceinline__ float bf2f(__bf16 v) {
    unsigned short u; __builtin_memcpy(&u, &v, 2);
    unsigned int w = ((unsigned int)u) << 16;
    float f; __builtin_memcpy(&f, &w, 4); return f;
}
static __device__ __forceinline__ __bf16 f2bf(float f) {
    unsigned int u; __builtin_memcpy(&u, &f, 4);
    unsigned int r = u + 0x7FFFu + ((u >> 16) & 1u);   // round-to-nearest-even
    unsigned short s = (unsigned short)(r >> 16);
    __bf16 b; __builtin_memcpy(&b, &s, 2); return b;
}
static __device__ __forceinline__ unsigned lds_off(const void* p) {
    // generic pointer to LDS: low 32 bits carry the LDS byte address
    return (unsigned)(unsigned long long)(uintptr_t)p;
}

// ---------------------------------------------------------------------------
// x[B,T,F] f32 -> h[B,T,C] bf16 (elementwise; layouts coincide)
// ---------------------------------------------------------------------------
__global__ void tcn_cvt_x(const float* __restrict__ x, __bf16* __restrict__ h) {
    long long i = (long long)blockIdx.x * 256 + threadIdx.x;
    if (i >= (long long)BN * TN * CN) return;
    h[i] = f2bf(x[i]);
}

// conv weights f32 [L][O][I][K] -> bf16 [L][K][O][I] (transposed for B-frags)
__global__ void tcn_cvt_w(const float* __restrict__ w1, const float* __restrict__ w2,
                          __bf16* __restrict__ o1, __bf16* __restrict__ o2) {
    int i = blockIdx.x * 256 + threadIdx.x;
    const int total = LN * CN * CN * KK;
    if (i >= total) return;
    int k  = i % KK;
    int r  = i / KK;
    int ic = r % CN; r /= CN;
    int oc = r % CN;
    int l  = r / CN;
    int dst = ((l * KK + k) * CN + oc) * CN + ic;
    o1[dst] = f2bf(w1[i]);
    o2[dst] = f2bf(w2[i]);
}

// ---------------------------------------------------------------------------
// Dilated causal conv as WMMA GEMM, activations in [B][T][C] bf16.
//   out[b,t,o] = act( sum_{c,k} in[b, t-(2-k)*D, c] * Wt[k][c][o] + bias[o] )
//   RES: out = relu(relu(conv+bias) + res)
// Block = 256 threads (8 wave32): 64(T) x 64(Cout) tile, one batch row.
//   D = A(time x cin) * B(cin x cout); tap shift = A row offset (free).
// ---------------------------------------------------------------------------
template <int D, bool RES>
__global__ __launch_bounds__(256) void tcn_conv(
        const __bf16* __restrict__ in, const __bf16* __restrict__ res,
        const __bf16* __restrict__ w,  const float* __restrict__ bias,
        __bf16* __restrict__ out) {
    constexpr int HALO = 2 * D;
    constexpr int ROWS = 64 + HALO;

    __shared__ __align__(16) __bf16 s_w[KK * CN * CN];   // [k][cout][cin]
    __shared__ __align__(16) __bf16 s_in[ROWS * CN];     // [t_local][c]

    const int tid  = threadIdx.x;
    const int b    = blockIdx.x >> 2;
    const int tile = blockIdx.x & 3;
    const int t0   = tile * 64;

    // stage weights: contiguous 16B-vector copy (L2-resident)
    {
        const uint4* wg = (const uint4*)w;
        uint4* sw = (uint4*)s_w;
        for (int i = tid; i < (KK * CN * CN) / 8; i += 256) sw[i] = wg[i];
    }

    // stage input tile rows [t0-HALO, t0+64) via async global->LDS b128;
    // out-of-range rows zero-filled with plain LDS stores
    {
        const unsigned sbase = lds_off(s_in);
        for (int idx = tid; idx < ROWS * 8; idx += 256) {
            const int r  = idx >> 3;
            const int ch = (idx & 7) * 8;
            const int t  = t0 - HALO + r;
            if (t >= 0 && t < TN) {
                const unsigned loff = sbase + (unsigned)(r * CN + ch) * 2;
                const unsigned long long ga =
                    (unsigned long long)(uintptr_t)&in[((long long)b * TN + t) * CN + ch];
                asm volatile("global_load_async_to_lds_b128 %0, %1, off"
                             :: "v"(loff), "v"(ga) : "memory");
            } else {
                uint4 z = {0u, 0u, 0u, 0u};
                *(uint4*)&s_in[r * CN + ch] = z;
            }
        }
    }
    if (tile < 3)  // hint next time-tile into cache
        __builtin_prefetch(&in[((long long)b * TN + t0 + 64) * CN], 0, 0);
    asm volatile("s_wait_asynccnt 0x0" ::: "memory");
    __syncthreads();

    const int lane = tid & 31;
    const int wave = tid >> 5;
    const int m    = wave & 3;            // time tile (M)
    const int n0   = (wave >> 2) * 2;     // cout tile pair {n0, n0+1} (N)
    const int lo16 = lane & 15;
    const int hi8  = (lane >> 4) << 3;    // 0 | 8
    const int hi16 = (lane >> 4) << 4;    // 0 | 16

    v8f acc0 = {};
    v8f acc1 = {};

    union Frag { v16bf v; uint4 q[2]; };

#pragma unroll
    for (int k = 0; k < KK; ++k) {
#pragma unroll
        for (int kc = 0; kc < 2; ++kc) {
            // A fragment: activation rows; tap shift folded into the row index
            const int t_loc = m * 16 + lo16 + k * D;
            const uint4* aq = (const uint4*)&s_in[t_loc * CN + kc * 32 + hi8];
            Frag fa; fa.q[0] = aq[0]; fa.q[1] = aq[2];

            // B fragments: Wt[k][cout-row][cin, same k order as A]
            const int crow = n0 * 16 + lo16;
            const uint4* bq = (const uint4*)&s_w[(k * CN + crow) * CN + kc * 32 + hi8];
            Frag fb0; fb0.q[0] = bq[0]; fb0.q[1] = bq[2];
            Frag fb1; fb1.q[0] = bq[128]; fb1.q[1] = bq[130];

            acc0 = __builtin_amdgcn_wmma_f32_16x16x32_bf16(
                       false, fa.v, false, fb0.v, (short)0, acc0, false, false);
            acc1 = __builtin_amdgcn_wmma_f32_16x16x32_bf16(
                       false, fa.v, false, fb1.v, (short)0, acc1, false, false);
        }
    }

    // epilogue: lane owns one output channel per acc; 8 consecutive t per acc
    const int oc0 = n0 * 16 + lo16;
    const int oc1 = oc0 + 16;
    const float bz0 = bias[oc0];
    const float bz1 = bias[oc1];
    const int tb = t0 + m * 16 + hi8;
#pragma unroll
    for (int r = 0; r < 8; ++r) {
        const int t = tb + r;
        if (t < TN) {
            const long long base = ((long long)b * TN + t) * CN;
            float v0 = fmaxf(acc0[r] + bz0, 0.0f);
            float v1 = fmaxf(acc1[r] + bz1, 0.0f);
            if (RES) {
                v0 = fmaxf(v0 + bf2f(res[base + oc0]), 0.0f);
                v1 = fmaxf(v1 + bf2f(res[base + oc1]), 0.0f);
            }
            out[base + oc0] = f2bf(v0);
            out[base + oc1] = f2bf(v1);
        }
    }
}

// ---------------------------------------------------------------------------
// Global max-pool over T: pooled[b,c] = max_t h[b,t,c]
// ---------------------------------------------------------------------------
__global__ void tcn_pool(const __bf16* __restrict__ h, float* __restrict__ pooled) {
    int i = blockIdx.x * 256 + threadIdx.x;
    if (i >= BN * CN) return;
    const int b = i >> 6;
    const int c = i & 63;
    const __bf16* p = h + (long long)b * TN * CN + c;
    float m = -INFINITY;
    for (int t = 0; t < TN; ++t) m = fmaxf(m, bf2f(p[(long long)t * CN]));
    pooled[i] = m;
}

// ---------------------------------------------------------------------------
// Head: z = relu(pooled @ proj^T + pb); attn = sigmoid(z@aw^T+ab); logit = z@fw^T+fb
// ---------------------------------------------------------------------------
__global__ __launch_bounds__(256) void tcn_head(
        const float* __restrict__ pooled,
        const float* __restrict__ pw, const float* __restrict__ pb,
        const float* __restrict__ aw, const float* __restrict__ ab,
        const float* __restrict__ fw, const float* __restrict__ fb,
        float* __restrict__ attn, float* __restrict__ logit) {
    __shared__ float zr[CN];
    const int b   = blockIdx.x;
    const int tid = threadIdx.x;
    if (tid < CN) {
        const float* pv = pooled + b * CN;
        const float* wr = pw + tid * CN;
        float acc = pb[tid];
#pragma unroll 8
        for (int c = 0; c < CN; ++c) acc += pv[c] * wr[c];
        zr[tid] = fmaxf(acc, 0.0f);
    }
    __syncthreads();
    for (int j = tid; j < NOUTN; j += 256) {
        const float* ar = aw + j * CN;
        const float* fr = fw + j * CN;
        float a = ab[j], f = fb[j];
#pragma unroll 8
        for (int c = 0; c < CN; ++c) { a += zr[c] * ar[c]; f += zr[c] * fr[c]; }
        attn[(long long)b * NOUTN + j]  = 1.0f / (1.0f + expf(-a));
        logit[(long long)b * NOUTN + j] = f;
    }
}

// ---------------------------------------------------------------------------
// block reductions (256 threads)
// ---------------------------------------------------------------------------
static __device__ __forceinline__ float blkSum(float v, float* red, int tid) {
    red[tid] = v; __syncthreads();
    for (int s = 128; s > 0; s >>= 1) {
        if (tid < s) red[tid] += red[tid + s];
        __syncthreads();
    }
    float r = red[0]; __syncthreads();
    return r;
}
static __device__ __forceinline__ float blkMax(float v, float* red, int tid) {
    red[tid] = v; __syncthreads();
    for (int s = 128; s > 0; s >>= 1) {
        if (tid < s) red[tid] = fmaxf(red[tid], red[tid + s]);
        __syncthreads();
    }
    float r = red[0]; __syncthreads();
    return r;
}

// ---------------------------------------------------------------------------
// Top-20 select + masked softmax + water-filling rebalance. One block per row.
// ---------------------------------------------------------------------------
__global__ __launch_bounds__(256) void tcn_select(
        const float* __restrict__ attn, const float* __restrict__ logit,
        float* __restrict__ out) {
    __shared__ float sv[NOUTN];
    __shared__ float sl[NOUTN];
    __shared__ float smask[NOUTN];
    __shared__ float red[256];
    __shared__ int   redi[256];
    __shared__ int   sflag, sdone;

    const int b   = blockIdx.x;
    const int tid = threadIdx.x;

    for (int j = tid; j < NOUTN; j += 256) {
        sv[j] = attn[(long long)b * NOUTN + j];
        smask[j] = 0.0f;
    }
    if (tid == 0) sdone = 0;
    __syncthreads();

    // ---- top-20 via iterative argmax (ties -> smallest index) ----
    for (int it = 0; it < NSELN; ++it) {
        float bv = -INFINITY; int bi = -1;
        for (int j = tid; j < NOUTN; j += 256)
            if (sv[j] > bv) { bv = sv[j]; bi = j; }
        red[tid] = bv; redi[tid] = bi;
        __syncthreads();
        for (int s = 128; s > 0; s >>= 1) {
            if (tid < s) {
                float ov = red[tid + s]; int oi = redi[tid + s];
                if (ov > red[tid] || (ov == red[tid] && oi != -1 &&
                                      (redi[tid] == -1 || oi < redi[tid]))) {
                    red[tid] = ov; redi[tid] = oi;
                }
            }
            __syncthreads();
        }
        if (tid == 0) { int w = redi[0]; smask[w] = 1.0f; sv[w] = -INFINITY; }
        __syncthreads();
    }

    // ---- softmax over logits, mask, renormalize ----
    float lm = -INFINITY;
    for (int j = tid; j < NOUTN; j += 256)
        lm = fmaxf(lm, logit[(long long)b * NOUTN + j]);
    const float mx = blkMax(lm, red, tid);

    float es = 0.0f;
    for (int j = tid; j < NOUTN; j += 256) {
        float e = expf(logit[(long long)b * NOUTN + j] - mx);
        sl[j] = e; es += e;
    }
    const float S = blkSum(es, red, tid);

    float qs = 0.0f;
    for (int j = tid; j < NOUTN; j += 256) {
        float q = (sl[j] / S) * smask[j];
        sl[j] = q; qs += q;
    }
    const float Q = blkSum(qs, red, tid);
    for (int j = tid; j < NOUTN; j += 256) sl[j] = sl[j] / (Q + 1e-8f);
    __syncthreads();

    // ---- water-filling: 1 initial + up to 16 scan steps, freeze on done ----
    for (int it = 0; it < 17; ++it) {
        __syncthreads();
        if (sdone) break;                         // uniform (post-barrier read)
        float dsum = 0.0f, nsum = 0.0f;
        for (int j = tid; j < NOUTN; j += 256) {
            float old = sl[j];
            float wc  = fminf(fmaxf(old, LBF), UBF);
            sv[j] = wc;
            dsum += old - wc;
            nsum += (wc != UBF) ? wc : 0.0f;
        }
        const float leftover = blkSum(dsum, red, tid);
        float ns = blkSum(nsum, red, tid);
        if (ns == 0.0f) ns = 1.0f;
        if (tid == 0) sflag = 0;
        __syncthreads();
        int over = 0;
        for (int j = tid; j < NOUTN; j += 256) {
            float wc = sv[j];
            float nv = (wc != UBF) ? wc + leftover * wc / ns : wc;
            sl[j] = nv;
            if (nv > UBF) over = 1;
        }
        if (over) sflag = 1;
        __syncthreads();
        if (tid == 0 && sflag == 0) sdone = 1;
    }
    __syncthreads();

    for (int j = tid; j < NOUTN; j += 256)
        out[(long long)b * NOUTN + j] = sl[j];
}

// ---------------------------------------------------------------------------
// launch
// ---------------------------------------------------------------------------
extern "C" void kernel_launch(void* const* d_in, const int* in_sizes, int n_in,
                              void* d_out, int out_size, void* d_ws, size_t ws_size,
                              hipStream_t stream) {
    const float* x   = (const float*)d_in[0];
    const float* c1w = (const float*)d_in[1];
    const float* c1b = (const float*)d_in[2];
    const float* c2w = (const float*)d_in[3];
    const float* c2b = (const float*)d_in[4];
    const float* pw  = (const float*)d_in[5];
    const float* pb  = (const float*)d_in[6];
    const float* aw  = (const float*)d_in[7];
    const float* ab  = (const float*)d_in[8];
    const float* fw  = (const float*)d_in[9];
    const float* fb  = (const float*)d_in[10];
    float* out = (float*)d_out;

    // workspace carve
    char* p = (char*)d_ws;
    auto alloc = [&](size_t bytes) -> void* {
        void* r = (void*)p;
        p += (bytes + 255) & ~(size_t)255;
        return r;
    };
    const size_t HB = (size_t)BN * TN * CN;                  // activation elems
    __bf16* hA   = (__bf16*)alloc(HB * 2);
    __bf16* hM   = (__bf16*)alloc(HB * 2);
    __bf16* hB   = (__bf16*)alloc(HB * 2);
    const size_t WE = (size_t)LN * KK * CN * CN;             // weight elems
    __bf16* wb1  = (__bf16*)alloc(WE * 2);
    __bf16* wb2  = (__bf16*)alloc(WE * 2);
    float* pooled = (float*)alloc((size_t)BN * CN * 4);
    float* attnB  = (float*)alloc((size_t)BN * NOUTN * 4);
    float* logitB = (float*)alloc((size_t)BN * NOUTN * 4);

    // weight + input conversion to bf16
    tcn_cvt_w<<<(int)((WE + 255) / 256), 256, 0, stream>>>(c1w, c2w, wb1, wb2);
    tcn_cvt_x<<<(int)((HB + 255) / 256), 256, 0, stream>>>(x, hA);

    // TCN levels: conv1 (relu) -> conv2 (relu + residual + relu)
    const int convGrid = BN * 4;                              // 4 time tiles of 64
    __bf16* cin  = hA;
    __bf16* cout = hB;
    for (int l = 0; l < LN; ++l) {
        const __bf16* wl1 = wb1 + (size_t)l * KK * CN * CN;
        const __bf16* wl2 = wb2 + (size_t)l * KK * CN * CN;
        const float*  bl1 = c1b + l * CN;
        const float*  bl2 = c2b + l * CN;
        switch (l) {
        case 0:
            tcn_conv<1, false><<<convGrid, 256, 0, stream>>>(cin, nullptr, wl1, bl1, hM);
            tcn_conv<1, true ><<<convGrid, 256, 0, stream>>>(hM, cin, wl2, bl2, cout);
            break;
        case 1:
            tcn_conv<2, false><<<convGrid, 256, 0, stream>>>(cin, nullptr, wl1, bl1, hM);
            tcn_conv<2, true ><<<convGrid, 256, 0, stream>>>(hM, cin, wl2, bl2, cout);
            break;
        case 2:
            tcn_conv<4, false><<<convGrid, 256, 0, stream>>>(cin, nullptr, wl1, bl1, hM);
            tcn_conv<4, true ><<<convGrid, 256, 0, stream>>>(hM, cin, wl2, bl2, cout);
            break;
        default:
            tcn_conv<8, false><<<convGrid, 256, 0, stream>>>(cin, nullptr, wl1, bl1, hM);
            tcn_conv<8, true ><<<convGrid, 256, 0, stream>>>(hM, cin, wl2, bl2, cout);
            break;
        }
        __bf16* t = cin; cin = cout; cout = t;
    }

    // head
    tcn_pool<<<(BN * CN + 255) / 256, 256, 0, stream>>>(cin, pooled);
    tcn_head<<<BN, 256, 0, stream>>>(pooled, pw, pb, aw, ab, fw, fb, attnB, logitB);
    tcn_select<<<BN, 256, 0, stream>>>(attnB, logitB, out);
}
